// ParallelSelfAttention_89524298318452
// MI455X (gfx1250) — compile-verified, matrix operand order checked by reading, not run on hardware
//
#include <hip/hip_runtime.h>
#include <hip/hip_bf16.h>

#define SEQ     2048
#define BATCHN  2
#define HIDDEN  2048
#define NHEADS  16
#define HDIM    128
#define MROWS   (SEQ * BATCHN)     // 4096
#define NQKV    (3 * HIDDEN)       // 6144
#define BHN     (BATCHN * NHEADS)  // 32
#define MASKVAL (-10000.0f)

typedef __attribute__((ext_vector_type(16))) __bf16 v16bf;
typedef __attribute__((ext_vector_type(8)))  __bf16 v8bf;
typedef __attribute__((ext_vector_type(8)))  float  v8f;

__device__ __forceinline__ unsigned short f32_to_bf16(float f) {
  union { float f; unsigned int u; } x; x.f = f;
  unsigned int u = x.u;
  unsigned int r = u + 0x7FFFu + ((u >> 16) & 1u);   // round-to-nearest-even
  return (unsigned short)(r >> 16);
}

__device__ __forceinline__ v16bf cat8(v8bf lo, v8bf hi) {
  return __builtin_shufflevector(lo, hi, 0,1,2,3,4,5,6,7,8,9,10,11,12,13,14,15);
}

// 16x32 bf16 A/B fragment from a row-major tile; rowptr already points at this
// lane's row start + K-block base. Per-lane 16B chunks at half*8 and 16+half*8
// (ISA 7.12.2 16-bit A-matrix striping; B mirrors with rows = N).
__device__ __forceinline__ v16bf load_frag(const unsigned short* rowptr, int half) {
  v8bf lo = *(const v8bf*)(rowptr + half * 8);
  v8bf hi = *(const v8bf*)(rowptr + 16 + half * 8);
  return cat8(lo, hi);
}

__device__ __forceinline__ v8f wmma_bf16(v16bf a, v16bf b, v8f c) {
  return __builtin_amdgcn_wmma_f32_16x16x32_bf16(false, a, false, b, (short)0, c,
                                                 false, false);
}

// ---------------- precision conversion ----------------

__global__ void cast_bf16_kernel(const float* __restrict__ src,
                                 unsigned short* __restrict__ dst, int n) {
  int i = blockIdx.x * blockDim.x + threadIdx.x;
  if (i < n) dst[i] = f32_to_bf16(src[i]);
}

// dst[c*R + r] = bf16(src[r*C + c])   (pre-transpose weights for contiguous B-frags)
__global__ void transpose_cast_kernel(const float* __restrict__ src,
                                      unsigned short* __restrict__ dst,
                                      int R, int C) {
  int i = blockIdx.x * blockDim.x + threadIdx.x;
  if (i < R * C) {
    int r = i / C, c = i - r * C;
    dst[(size_t)c * R + r] = f32_to_bf16(src[i]);
  }
}

// ---------------- shared 128x128 LDS-tiled GEMM core ----------------
// 8 waves; wave w computes rows [w*16, w*16+16) x 128 cols of the block tile.
// A: [M,K] row-major bf16; Bt: [N,K] row-major bf16 (i.e. B transposed).
// Per K-step stages a 128x32 panel of each into LDS (10 KB each, padded).

__device__ __forceinline__ void gemm_128x128_core(
    const unsigned short* __restrict__ A,
    const unsigned short* __restrict__ Bt,
    int K, int rowBase, int colBase,
    unsigned short (*As)[40], unsigned short (*Bs)[40],
    v8f acc[8]) {
  const int tid  = threadIdx.x;
  const int lane = tid & 31;
  const int wave = tid >> 5;           // 0..7
  const int half = lane >> 4;
  const int lidx = lane & 15;

#pragma unroll
  for (int t = 0; t < 8; ++t) acc[t] = (v8f){};

  const int r128 = tid & 127;          // staging row (uniform branch per wave)
  const unsigned short* srcBase =
      (tid < 128) ? (A + (size_t)(rowBase + r128) * K)
                  : (Bt + (size_t)(colBase + r128) * K);
  unsigned short(*dst)[40] = (tid < 128) ? As : Bs;

  for (int kb = 0; kb < K; kb += 32) {
    // cooperative stage: 128 rows x 32 bf16 (64B) per matrix, 1 row per thread
    const unsigned short* src = srcBase + kb;
    __builtin_prefetch(src + 64, 0, 0);
    v8bf c0 = *(const v8bf*)(src);
    v8bf c1 = *(const v8bf*)(src + 8);
    v8bf c2 = *(const v8bf*)(src + 16);
    v8bf c3 = *(const v8bf*)(src + 24);
    *(v8bf*)(&dst[r128][0])  = c0;
    *(v8bf*)(&dst[r128][8])  = c1;
    *(v8bf*)(&dst[r128][16]) = c2;
    *(v8bf*)(&dst[r128][24]) = c3;
    __syncthreads();

    v16bf a = load_frag(&As[wave * 16 + lidx][0], half);
#pragma unroll
    for (int t = 0; t < 8; ++t) {
      v16bf b = load_frag(&Bs[t * 16 + lidx][0], half);
      acc[t] = wmma_bf16(a, b, acc[t]);
    }
    __syncthreads();
  }
}

// ---------------- fused QKV projection ----------------
// C[4096,6144] = hb @ wqkv + b_qkv, scattered into q[bh][s][d], k[bh][t][d],
// vT[bh][d][t]  (bh = b*16 + head)

__global__ __launch_bounds__(256) void qkv_gemm_kernel(
    const unsigned short* __restrict__ hb,      // [4096,2048] bf16
    const unsigned short* __restrict__ wqkvT,   // [6144,2048] bf16 (transposed)
    const float* __restrict__ b_qkv,            // [6144]
    unsigned short* __restrict__ qmat,
    unsigned short* __restrict__ kmat,
    unsigned short* __restrict__ vT) {
  __shared__ unsigned short As[128][40];
  __shared__ unsigned short Bs[128][40];
  const int colBase = blockIdx.x * 128;   // 0..6016
  const int rowBase = blockIdx.y * 128;   // 0..3968
  const int lane = threadIdx.x & 31;
  const int wave = threadIdx.x >> 5;
  const int half = lane >> 4;
  const int lidx = lane & 15;

  v8f acc[8];
  gemm_128x128_core(hb, wqkvT, HIDDEN, rowBase, colBase, As, Bs, acc);

#pragma unroll
  for (int t = 0; t < 8; ++t) {
#pragma unroll
    for (int i = 0; i < 8; ++i) {
      int r = rowBase + wave * 16 + i + half * 8;   // flat row -> (s, b)
      int c = colBase + t * 16 + lidx;              // qkv column
      float v = acc[t][i] + b_qkv[c];
      int s = r >> 1, bidx = r & 1;
      int head = c / (3 * HDIM), j = c % (3 * HDIM);
      int bh = bidx * NHEADS + head;
      unsigned short bv = f32_to_bf16(v);
      if (j < HDIM)
        qmat[((size_t)bh * SEQ + s) * HDIM + j] = bv;
      else if (j < 2 * HDIM)
        kmat[((size_t)bh * SEQ + s) * HDIM + (j - HDIM)] = bv;
      else
        vT[((size_t)bh * HDIM + (j - 2 * HDIM)) * SEQ + s] = bv;
    }
  }
}

// ---------------- causal flash attention ----------------
// Block = 4 waves = 64 query rows of one (batch,head); K/V panels staged
// cooperatively in LDS, online softmax per wave, WMMA for S=QK^T and O+=PV.

__global__ __launch_bounds__(128) void attn_kernel(
    const unsigned short* __restrict__ qmat,
    const unsigned short* __restrict__ kmat,
    const unsigned short* __restrict__ vT,
    unsigned short* __restrict__ ctx) {
  __shared__ unsigned short Ks[32][136];      // 32 keys x 128 dims (padded)
  __shared__ unsigned short Vs[128][40];      // 128 dims x 32 keys (padded)
  __shared__ unsigned short pst[4][16][40];   // per-wave P transpose staging

  const int q0   = blockIdx.x * 64;
  const int bh   = blockIdx.y;
  const int tid  = threadIdx.x;
  const int lane = tid & 31;
  const int wave = tid >> 5;                  // 0..3
  const int half = lane >> 4;
  const int lidx = lane & 15;
  const int qw   = q0 + wave * 16;            // this wave's first query row
  const float inv_norm = 0.08838834764831845f;  // 1/sqrt(128)

  // Q fragments: 16 rows x 128 dims (4 K-chunks of 32)
  const unsigned short* qrow = qmat + ((size_t)bh * SEQ + qw + lidx) * HDIM;
  v16bf aq[4];
#pragma unroll
  for (int kc = 0; kc < 4; ++kc) aq[kc] = load_frag(qrow + kc * 32, half);

  v8f o[8];
#pragma unroll
  for (int dt = 0; dt < 8; ++dt) o[dt] = (v8f){};
  float mrow[8], lrow[8];
#pragma unroll
  for (int i = 0; i < 8; ++i) { mrow[i] = -3.0e38f; lrow[i] = 0.0f; }

  // staging indices (uniform per wave)
  const int krow = tid >> 2;                  // 0..31 key row
  const int kcol = (tid & 3) * 32;            // 0/32/64/96

  const int t_hi = q0 + 63;                   // block-wide causal bound
  for (int t0 = 0; t0 <= t_hi; t0 += 32) {
    // ---- cooperative stage of K panel (32x128) and V^T panel (128x32) ----
    {
      const unsigned short* ksrc =
          kmat + ((size_t)bh * SEQ + t0 + krow) * HDIM + kcol;
      v8bf k0 = *(const v8bf*)(ksrc);
      v8bf k1 = *(const v8bf*)(ksrc + 8);
      v8bf k2 = *(const v8bf*)(ksrc + 16);
      v8bf k3 = *(const v8bf*)(ksrc + 24);
      *(v8bf*)(&Ks[krow][kcol])      = k0;
      *(v8bf*)(&Ks[krow][kcol + 8])  = k1;
      *(v8bf*)(&Ks[krow][kcol + 16]) = k2;
      *(v8bf*)(&Ks[krow][kcol + 24]) = k3;
      const unsigned short* vsrc = vT + ((size_t)bh * HDIM + tid) * SEQ + t0;
      v8bf w0 = *(const v8bf*)(vsrc);
      v8bf w1 = *(const v8bf*)(vsrc + 8);
      v8bf w2 = *(const v8bf*)(vsrc + 16);
      v8bf w3 = *(const v8bf*)(vsrc + 24);
      *(v8bf*)(&Vs[tid][0])  = w0;
      *(v8bf*)(&Vs[tid][8])  = w1;
      *(v8bf*)(&Vs[tid][16]) = w2;
      *(v8bf*)(&Vs[tid][24]) = w3;
    }
    __syncthreads();

    // ---- scores: two 16x16 tiles over keys [t0, t0+32) ----
    v8f s0 = {}, s1 = {};
#pragma unroll
    for (int kc = 0; kc < 4; ++kc) {
      v16bf b0 = load_frag(&Ks[lidx][kc * 32], half);
      v16bf b1 = load_frag(&Ks[16 + lidx][kc * 32], half);
      s0 = wmma_bf16(aq[kc], b0, s0);
      s1 = wmma_bf16(aq[kc], b1, s1);
    }

    // ---- online softmax (rows live within 16-lane halves) ----
    const int ta = t0 + lidx;
    const int tb = t0 + 16 + lidx;
#pragma unroll
    for (int i = 0; i < 8; ++i) {
      int srow = qw + i + half * 8;
      float x0 = (ta <= srow) ? s0[i] * inv_norm : MASKVAL;
      float x1 = (tb <= srow) ? s1[i] * inv_norm : MASKVAL;
      float cm = fmaxf(x0, x1);
#pragma unroll
      for (int off = 1; off < 16; off <<= 1)
        cm = fmaxf(cm, __shfl_xor(cm, off, 32));
      float mn    = fmaxf(mrow[i], cm);
      float alpha = __expf(mrow[i] - mn);
      float p0 = __expf(x0 - mn);
      float p1 = __expf(x1 - mn);
      float rs = p0 + p1;
#pragma unroll
      for (int off = 1; off < 16; off <<= 1)
        rs += __shfl_xor(rs, off, 32);
      lrow[i] = lrow[i] * alpha + rs;
      mrow[i] = mn;
#pragma unroll
      for (int dt = 0; dt < 8; ++dt) o[dt][i] *= alpha;
      pst[wave][i + half * 8][lidx]      = f32_to_bf16(p0);
      pst[wave][i + half * 8][lidx + 16] = f32_to_bf16(p1);
    }
    asm volatile("s_wait_dscnt 0x0" ::: "memory");   // wave-local LDS RAW

    // ---- O += P @ V ----
    v16bf ap = load_frag(&pst[wave][lidx][0], half); // P as A-fragment
#pragma unroll
    for (int dt = 0; dt < 8; ++dt) {
      v16bf bv = load_frag(&Vs[dt * 16 + lidx][0], half);
      o[dt] = wmma_bf16(ap, bv, o[dt]);
    }
    __syncthreads();   // protect Ks/Vs before next stage
  }

  // ---- normalize + write context [row=(s*B+b), col=head*128+d] as bf16 ----
  const int bidx = bh >> 4, head = bh & 15;
#pragma unroll
  for (int dt = 0; dt < 8; ++dt) {
#pragma unroll
    for (int i = 0; i < 8; ++i) {
      int srow = qw + i + half * 8;
      float val = o[dt][i] / lrow[i];
      ctx[((size_t)(srow * BATCHN + bidx)) * HIDDEN + head * HDIM + dt * 16 + lidx] =
          f32_to_bf16(val);
    }
  }
}

// ---------------- dense output projection (f32 out, bias returned separately) --

__global__ __launch_bounds__(256) void dense_gemm_kernel(
    const unsigned short* __restrict__ ctx,   // [4096,2048] bf16
    const unsigned short* __restrict__ wdT,   // [2048,2048] bf16 (transposed)
    float* __restrict__ out) {
  __shared__ unsigned short As[128][40];
  __shared__ unsigned short Bs[128][40];
  const int colBase = blockIdx.x * 128;
  const int rowBase = blockIdx.y * 128;
  const int lane = threadIdx.x & 31;
  const int wave = threadIdx.x >> 5;
  const int half = lane >> 4;
  const int lidx = lane & 15;

  v8f acc[8];
  gemm_128x128_core(ctx, wdT, HIDDEN, rowBase, colBase, As, Bs, acc);

#pragma unroll
  for (int t = 0; t < 8; ++t) {
#pragma unroll
    for (int i = 0; i < 8; ++i) {
      int r = rowBase + wave * 16 + i + half * 8;
      int c = colBase + t * 16 + lidx;
      out[(size_t)r * HIDDEN + c] = acc[t][i];
    }
  }
}

// ---------------- host-side orchestration ----------------

extern "C" void kernel_launch(void* const* d_in, const int* in_sizes, int n_in,
                              void* d_out, int out_size, void* d_ws, size_t ws_size,
                              hipStream_t stream) {
  const float* hidden  = (const float*)d_in[0];   // [2048,2,2048]
  // d_in[1] = attention_mask (causal, handled analytically)
  const float* w_qkv   = (const float*)d_in[2];   // [2048,6144]
  const float* b_qkv   = (const float*)d_in[3];   // [6144]
  const float* w_dense = (const float*)d_in[4];   // [2048,2048]
  const float* b_dense = (const float*)d_in[5];   // [2048]
  float* out = (float*)d_out;

  // bf16 staging layout in d_ws (ushort elements)
  unsigned short* ws = (unsigned short*)d_ws;
  size_t off = 0;
  unsigned short* hb    = ws + off; off += (size_t)MROWS * HIDDEN;
  unsigned short* wqkvT = ws + off; off += (size_t)NQKV * HIDDEN;
  unsigned short* wdT   = ws + off; off += (size_t)HIDDEN * HIDDEN;
  unsigned short* qmat  = ws + off; off += (size_t)BHN * SEQ * HDIM;
  unsigned short* kmat  = ws + off; off += (size_t)BHN * SEQ * HDIM;
  unsigned short* vT    = ws + off; off += (size_t)BHN * HDIM * SEQ;
  unsigned short* ctx   = ws + off; off += (size_t)MROWS * HIDDEN;
  (void)ws_size; (void)in_sizes; (void)n_in; (void)out_size;

  // 1) precision conversion
  {
    int n = MROWS * HIDDEN;
    cast_bf16_kernel<<<(n + 255) / 256, 256, 0, stream>>>(hidden, hb, n);
  }
  {
    int n = HIDDEN * NQKV;
    transpose_cast_kernel<<<(n + 255) / 256, 256, 0, stream>>>(w_qkv, wqkvT,
                                                               HIDDEN, NQKV);
  }
  {
    int n = HIDDEN * HIDDEN;
    transpose_cast_kernel<<<(n + 255) / 256, 256, 0, stream>>>(w_dense, wdT,
                                                               HIDDEN, HIDDEN);
  }

  // 2) fused QKV projection (LDS-tiled WMMA GEMM, 128x128 per block)
  {
    dim3 grid(NQKV / 128, MROWS / 128);   // (48, 32)
    qkv_gemm_kernel<<<grid, 256, 0, stream>>>(hb, wqkvT, b_qkv, qmat, kmat, vT);
  }

  // 3) causal flash attention (LDS K/V panels + WMMA + online softmax)
  {
    dim3 grid(SEQ / 64, BHN);             // (32, 32)
    attn_kernel<<<grid, 128, 0, stream>>>(qmat, kmat, vT, ctx);
  }

  // 4) dense projection, f32 out
  {
    dim3 grid(HIDDEN / 128, MROWS / 128); // (16, 32)
    dense_gemm_kernel<<<grid, 256, 0, stream>>>(ctx, wdT, out);
  }

  // 5) tuple tail: bias returned separately (skip_bias_add)
  hipMemcpyAsync(out + (size_t)MROWS * HIDDEN, b_dense, HIDDEN * sizeof(float),
                 hipMemcpyDeviceToDevice, stream);
}
